// DiGCN_12833362280699
// MI455X (gfx1250) — matrix operand order repaired and checked by reading, not compiled, not run on hardware
//
#include <hip/hip_runtime.h>
#include <stdint.h>

typedef __attribute__((ext_vector_type(16))) __bf16 v16bf;
typedef __attribute__((ext_vector_type(8)))  __bf16 v8bf;
typedef __attribute__((ext_vector_type(8)))  float  v8f;

#define FD 128  // feature dim

// async 16B copy: global (SADDR + voff + imm) -> LDS (ldsaddr + imm)
// GVS mode: INST_OFFSET is added to BOTH global and LDS addresses (ISA §15.18).
#define ASYNC_CP16(ldsaddr, voff, sbase, IMM)                                 \
  asm volatile("global_load_async_to_lds_b128 %0, %1, %2 offset:" IMM        \
               :: "v"(ldsaddr), "v"(voff), "s"(sbase) : "memory")

// ---------------- utility kernels ----------------

__global__ void k_zero(float* __restrict__ p, long n) {
  long i = (long)blockIdx.x * blockDim.x + threadIdx.x;
  long s = (long)gridDim.x * blockDim.x;
  for (; i < n; i += s) p[i] = 0.0f;
}

__global__ void k_f2b(const float* __restrict__ src, __bf16* __restrict__ dst, long n) {
  long i = (long)blockIdx.x * blockDim.x + threadIdx.x;
  long s = (long)gridDim.x * blockDim.x;
  for (; i < n; i += s) dst[i] = (__bf16)src[i];
}

// deg[c] += w[e]  (targets)
__global__ void k_deg(const long long* __restrict__ ei, const float* __restrict__ ew,
                      float* __restrict__ deg, int E) {
  int e = blockIdx.x * blockDim.x + threadIdx.x;
  if (e < E) atomicAdd(&deg[(long)ei[(long)E + e]], ew[e]);
}

// norm[e] = dinv(row) * w * dinv(col)
__global__ void k_edge_norm(const long long* __restrict__ ei, const float* __restrict__ ew,
                            const float* __restrict__ deg, float* __restrict__ nrm, int E) {
  int e = blockIdx.x * blockDim.x + threadIdx.x;
  if (e < E) {
    long r = (long)ei[e];
    long c = (long)ei[(long)E + e];
    float dr = deg[r], dc = deg[c];
    float ir = dr > 0.0f ? rsqrtf(dr) : 0.0f;
    float ic = dc > 0.0f ? rsqrtf(dc) : 0.0f;
    nrm[e] = ir * ew[e] * ic;
  }
}

// ---------------- fused dual-branch WMMA GEMM ----------------
// hl = xb @ Wlin^T ; hg = xb @ Wgcn^T   (per-layer weights, row-major [n][k], bf16)
// Block: 256 threads = 8 waves; block computes 128 rows x 128 cols for both branches.
// Weights staged into LDS once per block via async global->LDS copies.
__global__ __launch_bounds__(256) void k_gemm_dual(
    const __bf16* __restrict__ xb,
    const __bf16* __restrict__ wlin,
    const __bf16* __restrict__ wgcn,
    float* __restrict__ hl, float* __restrict__ hg, int nrows) {
  __shared__ __bf16 smem[2 * FD * FD];  // [0,16384): Wlin  [16384,32768): Wgcn  (64KB)

  const int tid  = threadIdx.x;
  const int wave = tid >> 5;           // 0..7 -> 16-row strip
  const int lane = tid & 31;
  const int half = lane >> 4;          // 0/1
  const int l16  = lane & 15;

  // ---- async stage 2 x 32KB of weights into LDS ----
  {
    const uint32_t lbase = (uint32_t)(uintptr_t)(void*)smem;
    const uint32_t voff  = (uint32_t)tid * 16u;       // per-thread 16B slot in 4KB slab
    const uint32_t ldsL  = lbase + voff;
    const uint32_t ldsG  = lbase + 32768u + voff;
    const uint64_t sl = (uint64_t)(uintptr_t)wlin;
    const uint64_t sg = (uint64_t)(uintptr_t)wgcn;
    ASYNC_CP16(ldsL, voff, sl, "0");
    ASYNC_CP16(ldsL, voff, sl, "4096");
    ASYNC_CP16(ldsL, voff, sl, "8192");
    ASYNC_CP16(ldsL, voff, sl, "12288");
    ASYNC_CP16(ldsL, voff, sl, "16384");
    ASYNC_CP16(ldsL, voff, sl, "20480");
    ASYNC_CP16(ldsL, voff, sl, "24576");
    ASYNC_CP16(ldsL, voff, sl, "28672");
    ASYNC_CP16(ldsG, voff, sg, "0");
    ASYNC_CP16(ldsG, voff, sg, "4096");
    ASYNC_CP16(ldsG, voff, sg, "8192");
    ASYNC_CP16(ldsG, voff, sg, "12288");
    ASYNC_CP16(ldsG, voff, sg, "16384");
    ASYNC_CP16(ldsG, voff, sg, "20480");
    ASYNC_CP16(ldsG, voff, sg, "24576");
    ASYNC_CP16(ldsG, voff, sg, "28672");
    asm volatile("s_wait_asynccnt 0x0" ::: "memory");  // this wave's copies done
  }
  __syncthreads();                                     // LDS visible to all waves

  const long row0 = (long)blockIdx.x * 128 + wave * 16;

  v8f accL[8], accG[8];
#pragma unroll
  for (int t = 0; t < 8; ++t) { accL[t] = (v8f){}; accG[t] = (v8f){}; }

  const long arow = row0 + l16;
  const bool rvalid = arow < (long)nrows;
  const __bf16* aptr = xb + arow * FD;

#pragma unroll
  for (int k0 = 0; k0 < FD; k0 += 32) {
    // ---- A fragment: 16x32 bf16, contiguous k = 16h + s, same order as B ----
    v16bf afrag;
    if (rvalid) {
      v8bf c0 = *(const v8bf*)(aptr + k0 + 16 * half);       // k = k0+16h .. +7
      v8bf c1 = *(const v8bf*)(aptr + k0 + 16 * half + 8);   // k = k0+16h+8 .. +7
#pragma unroll
      for (int j = 0; j < 8; ++j) { afrag[j] = c0[j]; afrag[j + 8] = c1[j]; }
    } else {
#pragma unroll
      for (int j = 0; j < 16; ++j) afrag[j] = (__bf16)0.0f;
    }
#pragma unroll
    for (int t = 0; t < 8; ++t) {
      const int n = t * 16 + l16;
      // ---- B fragment: 32x16, B[k][n] = W[n][k]; lane reads 16 contiguous bf16 from LDS ----
      {
        const __bf16* bp = smem + n * FD + k0 + 16 * half;   // Wlin in LDS
        v8bf b0 = *(const v8bf*)(bp);
        v8bf b1 = *(const v8bf*)(bp + 8);
        v16bf bfrag;
#pragma unroll
        for (int j = 0; j < 8; ++j) { bfrag[j] = b0[j]; bfrag[j + 8] = b1[j]; }
        accL[t] = __builtin_amdgcn_wmma_f32_16x16x32_bf16(
            false, afrag, false, bfrag, (short)0, accL[t], false, false);
      }
      {
        const __bf16* bp = smem + 16384 + n * FD + k0 + 16 * half;  // Wgcn in LDS
        v8bf b0 = *(const v8bf*)(bp);
        v8bf b1 = *(const v8bf*)(bp + 8);
        v16bf bfrag;
#pragma unroll
        for (int j = 0; j < 8; ++j) { bfrag[j] = b0[j]; bfrag[j + 8] = b1[j]; }
        accG[t] = __builtin_amdgcn_wmma_f32_16x16x32_bf16(
            false, afrag, false, bfrag, (short)0, accG[t], false, false);
      }
    }
  }

  // ---- store D: element r -> row = r + 8*half, col = t*16 + l16 ----
#pragma unroll
  for (int t = 0; t < 8; ++t) {
    const int n = t * 16 + l16;
#pragma unroll
    for (int r = 0; r < 8; ++r) {
      const long m = row0 + r + 8 * half;
      if (m < (long)nrows) {
        hl[m * FD + n] = accL[t][r];
        hg[m * FD + n] = accG[t][r];
      }
    }
  }
}

// ---------------- edge scatter: hr[col] += norm[e] * hg[row] ----------------
// 32 lanes per edge, 4 floats per lane (128 feats)
__global__ __launch_bounds__(256) void k_scatter(
    const long long* __restrict__ ei, const float* __restrict__ nrm,
    const float* __restrict__ hg, float* __restrict__ hr, int E) {
  const int lane = threadIdx.x & 31;
  const long e = ((long)blockIdx.x * blockDim.x + threadIdx.x) >> 5;
  if (e >= (long)E) return;
  const long r = (long)ei[e];
  const long c = (long)ei[(long)E + e];
  const float w = nrm[e];
  const float4 v = *((const float4*)(hg + r * FD) + lane);
  float* dst = hr + c * FD + lane * 4;
  atomicAdd(dst + 0, v.x * w);
  atomicAdd(dst + 1, v.y * w);
  atomicAdd(dst + 2, v.z * w);
  atomicAdd(dst + 3, v.w * w);
}

// ---------------- BN column stats: sums[c] += h, sums[128+c] += h*h ----------------
__global__ __launch_bounds__(128) void k_stats(
    const float* __restrict__ hl, const float* __restrict__ hr,
    float* __restrict__ sums, int n) {
  const int c = threadIdx.x;  // 0..127
  float s = 0.0f, s2 = 0.0f;
  for (long r = blockIdx.x; r < (long)n; r += gridDim.x) {
    float h = hl[r * FD + c] + hr[r * FD + c];
    s += h;
    s2 += h * h;
  }
  atomicAdd(&sums[c], s);
  atomicAdd(&sums[FD + c], s2);
}

__global__ void k_bn_coef(const float* __restrict__ sums,
                          const float* __restrict__ gamma, const float* __restrict__ beta,
                          float* __restrict__ coef, float inv_n) {
  const int c = threadIdx.x;
  if (c < FD) {
    float mu   = sums[c] * inv_n;
    float var  = sums[FD + c] * inv_n - mu * mu;
    float rstd = rsqrtf(var + 1e-5f);
    float a = gamma[c] * rstd;
    coef[c]      = a;
    coef[FD + c] = beta[c] - mu * a;
  }
}

// ---------------- apply BN (+ReLU), write f32 out and/or bf16 next-x ----------------
__global__ void k_apply(const float* __restrict__ hl, const float* __restrict__ hr,
                        const float* __restrict__ coef,
                        float* __restrict__ outf, __bf16* __restrict__ outb,
                        int relu, long total) {
  long i = (long)blockIdx.x * blockDim.x + threadIdx.x;
  long s = (long)gridDim.x * blockDim.x;
  for (; i < total; i += s) {
    int c = (int)(i & (FD - 1));
    float v = (hl[i] + hr[i]) * coef[c] + coef[FD + c];
    if (relu) v = fmaxf(v, 0.0f);
    if (outf) outf[i] = v;
    if (outb) outb[i] = (__bf16)v;
  }
}

// ---------------- host launcher ----------------
extern "C" void kernel_launch(void* const* d_in, const int* in_sizes, int n_in,
                              void* d_out, int out_size, void* d_ws, size_t ws_size,
                              hipStream_t stream) {
  const float*     x     = (const float*)d_in[0];
  const long long* ei    = (const long long*)d_in[1];   // int64 (2, E)
  const float*     ew    = (const float*)d_in[2];
  const float*     Wlin  = (const float*)d_in[3];       // (2,128,128)
  const float*     Wgcn  = (const float*)d_in[4];       // (2,128,128)
  const float*     gamma = (const float*)d_in[5];       // (2,128)
  const float*     beta  = (const float*)d_in[6];       // (2,128)
  float*           out   = (float*)d_out;

  const int N = in_sizes[0] / FD;
  const int E = in_sizes[2];
  const long ND = (long)N * FD;

  // workspace carve-up (256B aligned)
  char* wp = (char*)d_ws;
  auto alloc = [&](size_t bytes) -> void* {
    void* p = (void*)wp;
    wp += (bytes + 255) & ~(size_t)255;
    return p;
  };
  __bf16* xb   = (__bf16*)alloc((size_t)ND * 2);           // bf16 activations
  float*  hl   = (float*) alloc((size_t)ND * 4);
  float*  hg   = (float*) alloc((size_t)ND * 4);
  float*  hr   = (float*) alloc((size_t)ND * 4);
  float*  deg  = (float*) alloc((size_t)N * 4);
  float*  nrm  = (float*) alloc((size_t)E * 4);
  __bf16* WbL  = (__bf16*)alloc((size_t)2 * FD * FD * 2);
  __bf16* WbG  = (__bf16*)alloc((size_t)2 * FD * FD * 2);
  float*  sums = (float*) alloc(2 * FD * 4);
  float*  coef = (float*) alloc(2 * FD * 4);

  const int eb = (E + 255) / 256;

  // degree + symmetric normalization
  k_zero<<<(N + 255) / 256, 256, 0, stream>>>(deg, (long)N);
  k_deg<<<eb, 256, 0, stream>>>(ei, ew, deg, E);
  k_edge_norm<<<eb, 256, 0, stream>>>(ei, ew, deg, nrm, E);

  // bf16 conversions
  k_f2b<<<4096, 256, 0, stream>>>(x, xb, ND);
  k_f2b<<<64, 256, 0, stream>>>(Wlin, WbL, (long)2 * FD * FD);
  k_f2b<<<64, 256, 0, stream>>>(Wgcn, WbG, (long)2 * FD * FD);

  const int gemm_blocks    = (N + 127) / 128;
  const int scatter_blocks = (int)(((long)E * 32 + 255) / 256);
  const float inv_n = 1.0f / (float)N;

  for (int layer = 0; layer < 2; ++layer) {
    const __bf16* wl = WbL + (size_t)layer * FD * FD;
    const __bf16* wg = WbG + (size_t)layer * FD * FD;

    // fused dual GEMM: hl = x@Wlin^T, hg = x@Wgcn^T
    k_gemm_dual<<<gemm_blocks, 256, 0, stream>>>(xb, wl, wg, hl, hg, N);

    // hr = segment_sum(norm * hg[row]) over targets
    k_zero<<<4096, 256, 0, stream>>>(hr, ND);
    k_scatter<<<scatter_blocks, 256, 0, stream>>>(ei, nrm, hg, hr, E);

    // batch-norm statistics over nodes
    k_zero<<<1, 256, 0, stream>>>(sums, (long)(2 * FD));
    k_stats<<<1024, 128, 0, stream>>>(hl, hr, sums, N);
    k_bn_coef<<<1, 128, 0, stream>>>(sums, gamma + (size_t)layer * FD,
                                     beta + (size_t)layer * FD, coef, inv_n);

    if (layer == 0) {
      // ReLU, feed next layer as bf16
      k_apply<<<4096, 256, 0, stream>>>(hl, hr, coef, (float*)nullptr, xb, 1, ND);
    } else {
      // final: no ReLU, f32 output
      k_apply<<<4096, 256, 0, stream>>>(hl, hr, coef, out, (__bf16*)nullptr, 0, ND);
    }
  }
}
